// MemoryAgent_73495480369796
// MI455X (gfx1250) — matrix operand order repaired and checked by reading, not compiled, not run on hardware
//
#include <hip/hip_runtime.h>
#include <hip/hip_bf16.h>
#include <cstdint>
#include <cstddef>

// ---------------- problem dims ----------------
constexpr int TT  = 512;            // timesteps
constexpr int BB  = 256;            // batch
constexpr int FF  = 512;            // features
constexpr int HH  = 128;            // hidden
constexpr int AA  = 8;              // actions
constexpr int MHH = 64;             // mlp hidden
constexpr int G4H = 4 * HH;         // 512 gate columns
constexpr long TBROWS = (long)TT * BB;  // 131072

typedef __attribute__((ext_vector_type(16))) __bf16 v16bf;
typedef __attribute__((ext_vector_type(8)))  float  v8f;

union B16x8 { uint4 u; __bf16 h[8]; };
union V8F   { v8f v; float4 f[2]; };

__device__ __forceinline__ v8f wmma_bf16(v16bf a, v16bf b, v8f c) {
  // D(f32 16x16) = A(16x32 bf16) * B(32x16 bf16) + C
  return __builtin_amdgcn_wmma_f32_16x16x32_bf16(false, a, false, b,
                                                 short(0), c, false, false);
}

// ---- fast activations: native v_exp_f32 / v_rcp_f32 / v_tanh_f32 ----
__device__ __forceinline__ float fexp(float x) {
  return __builtin_amdgcn_exp2f(x * 1.442695040888963f);
}
__device__ __forceinline__ float frcp(float x) {
  return __builtin_amdgcn_rcpf(x);
}
__device__ __forceinline__ float fsigmoid(float x) {
  return frcp(1.0f + fexp(-x));
}
__device__ __forceinline__ float ftanh(float x) {
#if __has_builtin(__builtin_amdgcn_tanhf)
  return __builtin_amdgcn_tanhf(x);
#else
  float e = fexp(-2.0f * fabsf(x));            // in (0,1], overflow-safe
  float t = (1.0f - e) * frcp(1.0f + e);
  return copysignf(t, x);
#endif
}

// A fragment (16x32 bf16, M = lane&15 for both lane halves):
// element i -> K = kb + (i<8 ? 0 : 16) + hi*8 + (i&7); two 16B runs.
__device__ __forceinline__ v16bf make_afrag(const __bf16* prow, int kb, int hi) {
  B16x8 lo, hj;
  lo.u = *(const uint4*)(prow + kb + hi * 8);
  hj.u = *(const uint4*)(prow + kb + 16 + hi * 8);
  v16bf r;
#pragma unroll
  for (int i = 0; i < 8; i++) { r[i] = lo.h[i]; r[i + 8] = hj.h[i]; }
  return r;
}

// B fragment (32x16 bf16, N = lane&15):
// element i -> K = kb + (i<8 ? 0 : 16) + hi*8 + (i&7); two 16B runs.
__device__ __forceinline__ v16bf make_bfrag_bf16(const __bf16* prow, int kb, int hi) {
  B16x8 a, b;
  a.u = *(const uint4*)(prow + kb + hi * 8);
  b.u = *(const uint4*)(prow + kb + 16 + hi * 8);
  v16bf r;
#pragma unroll
  for (int i = 0; i < 8; i++) { r[i] = a.h[i]; r[i + 8] = b.h[i]; }
  return r;
}

// ================== Phase 0: one-time weight conversion =======================
// W_ih, W_hh -> bf16 (same layout); W_p1, W_v1 -> transposed bf16 [MH][H].
__global__ void __launch_bounds__(256) prep_weights_kernel(
    const float* __restrict__ W_ih, const float* __restrict__ W_hh,
    const float* __restrict__ W_p1, const float* __restrict__ W_v1,
    __bf16* __restrict__ W_ihB, __bf16* __restrict__ W_hhB,
    __bf16* __restrict__ Wp1T,  __bf16* __restrict__ Wv1T)
{
  long i0 = (long)blockIdx.x * blockDim.x + threadIdx.x;
  long st = (long)gridDim.x * blockDim.x;
  for (long i = i0; i < (long)G4H * FF; i += st) {
    W_ihB[i] = (__bf16)W_ih[i];
    if (i < (long)G4H * HH) W_hhB[i] = (__bf16)W_hh[i];
    if (i < (long)MHH * HH) {
      int nn = (int)(i >> 7), k = (int)(i & 127);   // i = nn*128 + k
      Wp1T[i] = (__bf16)W_p1[(long)k * MHH + nn];
      Wv1T[i] = (__bf16)W_v1[(long)k * MHH + nn];
    }
  }
}

// ================== Phase 1: Xg = x @ W_ih^T + (b_ih + b_hh) ==================
// Block = one 32-row tile (two 16-row WMMA tiles share every B fragment -> half
// the L2 weight traffic). 8 waves cover 64 gate cols each. Output stored in
// C-fragment block layout: Xg2[(rowtile16*32 + coltile)*256 + lane*8 + r].
constexpr int XST = 520;  // padded bf16 row stride for x tile in LDS
constexpr int MR  = 32;   // rows per block

__global__ void __launch_bounds__(256) gates_gemm_kernel(
    const float* __restrict__ x, const __bf16* __restrict__ W_ihB,
    const float* __restrict__ b_ih, const float* __restrict__ b_hh,
    float* __restrict__ Xg2)
{
  __shared__ __align__(16) __bf16 xsh[MR][XST];
  const int tid  = threadIdx.x, lane = tid & 31, w = tid >> 5;
  const int n    = lane & 15,   hi   = lane >> 4;
  const long row0 = (long)blockIdx.x * MR;

  // Cooperative stage of the 32x512 x tile (fp32 -> bf16), read once from HBM.
#pragma unroll
  for (int i = 0; i < 16; i++) {
    int idx = tid + i * 256;            // 4096 float4s total
    int row = idx >> 7;
    int c4  = (idx & 127) << 2;
    float4 v = *(const float4*)(x + (row0 + row) * FF + c4);
    xsh[row][c4 + 0] = (__bf16)v.x; xsh[row][c4 + 1] = (__bf16)v.y;
    xsh[row][c4 + 2] = (__bf16)v.z; xsh[row][c4 + 3] = (__bf16)v.w;
  }
  __syncthreads();

  const int nc0 = w * 64;
  v8f acc[2][4];
#pragma unroll
  for (int nt = 0; nt < 4; nt++) {
    float bv = b_ih[nc0 + nt * 16 + n] + b_hh[nc0 + nt * 16 + n];
#pragma unroll
    for (int r = 0; r < 8; r++) { acc[0][nt][r] = bv; acc[1][nt][r] = bv; }
  }

#pragma unroll 4
  for (int kc = 0; kc < 16; kc++) {
    v16bf a0 = make_afrag(&xsh[n][0],      kc * 32, hi);
    v16bf a1 = make_afrag(&xsh[16 + n][0], kc * 32, hi);
#pragma unroll
    for (int nt = 0; nt < 4; nt++) {
      const __bf16* p = W_ihB + (long)(nc0 + nt * 16 + n) * FF;  // row of W_ih
      v16bf b = make_bfrag_bf16(p, kc * 32, hi);
      acc[0][nt] = wmma_bf16(a0, b, acc[0][nt]);
      acc[1][nt] = wmma_bf16(a1, b, acc[1][nt]);
    }
  }

  // Coalesced fragment-block stores: 16-row tile index = blockIdx*2 + m2.
#pragma unroll
  for (int m2 = 0; m2 < 2; m2++)
#pragma unroll
    for (int nt = 0; nt < 4; nt++) {
      V8F u; u.v = acc[m2][nt];
      float4* dst = (float4*)(
          Xg2 + (((long)blockIdx.x * 2 + m2) * 32 + 4 * w + nt) * 256 + lane * 8);
      dst[0] = u.f[0];
      dst[1] = u.f[1];
    }
}

// ================== Phase 2: recurrent LSTM (batch-sliced, serial in T) =======
// 16 blocks x 256 threads. Block owns batch rows [slab*16, slab*16+16) for all T.
// Wave w owns hidden cols [16w,16w+16) -> gate tiles at coltile j = 8g + w.
// W_hh fragments live in registers for the whole sequence; h ping-pongs in LDS.
constexpr int HST = 136;  // padded bf16 row stride for h in LDS

__global__ void __launch_bounds__(256, 1) lstm_recurrent_kernel(
    const float* __restrict__ Xg2,  const float* __restrict__ done,
    const float* __restrict__ h0,   const float* __restrict__ c0,
    const __bf16* __restrict__ W_hhB, __bf16* __restrict__ hseqB,
    float* __restrict__ hT, float* __restrict__ cT)
{
  __shared__ __align__(16) __bf16 hbuf[2][16][HST];
  const int tid  = threadIdx.x, lane = tid & 31, w = tid >> 5;
  const int n    = lane & 15,   hi   = lane >> 4;
  const int slab = blockIdx.x;

  // Preload all 16 W_hh B-fragments (loop-invariant across T), already bf16.
  v16bf Bw[4][4];
#pragma unroll
  for (int kc = 0; kc < 4; kc++)
#pragma unroll
    for (int g = 0; g < 4; g++)
      Bw[kc][g] = make_bfrag_bf16(W_hhB + (long)(g * 128 + w * 16 + n) * HH,
                                  kc * 32, hi);

  // Init c registers (C/D layout: element r -> row r+8*hi) with done[0] mask.
  v8f creg, hreg;
#pragma unroll
  for (int r = 0; r < 8; r++) {
    int brow = slab * 16 + r + 8 * hi;
    float mm = 1.0f - done[brow];
    creg[r]  = c0[(long)brow * HH + w * 16 + n] * mm;
    hreg[r]  = 0.0f;
  }
  // Init LDS h with masked h0 (2048 elems, 8 per thread).
#pragma unroll
  for (int i = 0; i < 8; i++) {
    int idx = tid * 8 + i;
    int row = idx >> 7, col = idx & 127;
    int brow = slab * 16 + row;
    float mm = 1.0f - done[brow];
    hbuf[0][row][col] = (__bf16)(h0[(long)brow * HH + col] * mm);
  }
  int cur = 0;
  __syncthreads();

#pragma unroll 1
  for (int t = 0; t < TT; t++) {
    // A fragments of h (rows = batch, K = hidden) from LDS.
    v16bf Af[4];
#pragma unroll
    for (int kc = 0; kc < 4; kc++)
      Af[kc] = make_afrag(&hbuf[cur][n][0], kc * 32, hi);

    // Gate pre-activations from phase 1, fragment-block layout: 2 b128 per gate.
    v8f xgv[4];
#pragma unroll
    for (int g = 0; g < 4; g++) {
      const float4* p = (const float4*)(
          Xg2 + (((long)t * 16 + slab) * 32 + 8 * g + w) * 256 + lane * 8);
      V8F u; u.f[0] = p[0]; u.f[1] = p[1];
      xgv[g] = u.v;
    }
    if (t + 1 < TT) {
      // Prefetch next step's gate blocks (global_prefetch_b8) on the serial path.
#pragma unroll
      for (int g = 0; g < 4; g++)
        __builtin_prefetch(
            Xg2 + (((long)(t + 1) * 16 + slab) * 32 + 8 * g + w) * 256 + lane * 8,
            0, 1);
    }

    v8f acc[4];
#pragma unroll
    for (int g = 0; g < 4; g++) {
      v8f a = xgv[g];
#pragma unroll
      for (int kc = 0; kc < 4; kc++)
        a = wmma_bf16(Af[kc], Bw[kc][g], a);
      acc[g] = a;
    }

    // Next step's done mask: rows r+8*hi are 8 consecutive floats.
    v8f mnext;
    if (t + 1 < TT) {
      const float4* dp =
          (const float4*)(done + (long)(t + 1) * BB + slab * 16 + 8 * hi);
      float4 d0 = dp[0], d1 = dp[1];
      mnext[0] = 1.0f - d0.x; mnext[1] = 1.0f - d0.y;
      mnext[2] = 1.0f - d0.z; mnext[3] = 1.0f - d0.w;
      mnext[4] = 1.0f - d1.x; mnext[5] = 1.0f - d1.y;
      mnext[6] = 1.0f - d1.z; mnext[7] = 1.0f - d1.w;
    } else {
#pragma unroll
      for (int r = 0; r < 8; r++) mnext[r] = 1.0f;
    }

    B16x8 hb;
#pragma unroll
    for (int r = 0; r < 8; r++) {
      float iv = fsigmoid(acc[0][r]);
      float fv = fsigmoid(acc[1][r]);
      float gv = ftanh(acc[2][r]);
      float ov = fsigmoid(acc[3][r]);
      float c  = fv * creg[r] + iv * gv;
      float h  = ov * ftanh(c);
      hreg[r]  = h;
      hb.h[r]  = (__bf16)h;
      creg[r]  = c * mnext[r];
      hbuf[cur ^ 1][r + 8 * hi][w * 16 + n] = (__bf16)(h * mnext[r]);
    }
    // hseq as bf16 fragment blocks: one b128 store per lane per step.
    *(uint4*)(hseqB + (((long)t * 16 + slab) * 8 + w) * 256 + lane * 8) = hb.u;

    cur ^= 1;
    __syncthreads();
  }

#pragma unroll
  for (int r = 0; r < 8; r++) {
    int brow = slab * 16 + r + 8 * hi;
    hT[(long)brow * HH + w * 16 + n] = hreg[r];
    cT[(long)brow * HH + w * 16 + n] = creg[r];
  }
}

// ================== Phase 3: MLP heads over all T*B rows ======================
// Block = 8 independent waves, each owning one 16-row tile. h tile is rebuilt in
// per-wave LDS from phase-2 fragment blocks; stage-1 GEMMs use WMMA with bf16
// B-fragments straight from the pre-transposed weights; stage 2 is per-lane dots.
constexpr int HST3 = 136;  // padded bf16 row stride
constexpr int PST  = 65;   // padded fp32 activation stride

__global__ void __launch_bounds__(256) heads_kernel(
    const __bf16* __restrict__ hseqB,
    const __bf16* __restrict__ Wp1T, const float* __restrict__ b_p1,
    const float* __restrict__ W_p2,  const float* __restrict__ b_p2,
    const __bf16* __restrict__ Wv1T, const float* __restrict__ b_v1,
    const float* __restrict__ W_v2,  const float* __restrict__ b_v2,
    float* __restrict__ out)
{
  __shared__ __align__(16) __bf16 hsh[8][16][HST3];
  __shared__ float act[8][2][16][PST];   // [wave][head][row][k]

  const int tid = threadIdx.x, lane = tid & 31, w = tid >> 5;
  const int n   = lane & 15,   hi   = lane >> 4;
  const long tile  = (long)blockIdx.x * 8 + w;   // 0..8191
  const long grow0 = tile * 16;

  // Rebuild the 16x128 h tile from 8 fragment blocks (coalesced b128 loads).
#pragma unroll
  for (int w2 = 0; w2 < 8; w2++) {
    B16x8 hb;
    hb.u = *(const uint4*)(hseqB + (tile * 8 + w2) * 256 + lane * 8);
#pragma unroll
    for (int r = 0; r < 8; r++)
      hsh[w][r + 8 * hi][w2 * 16 + n] = hb.h[r];
  }

  v16bf Af[4];
#pragma unroll
  for (int kc = 0; kc < 4; kc++)
    Af[kc] = make_afrag(&hsh[w][n][0], kc * 32, hi);

#pragma unroll
  for (int head = 0; head < 2; head++) {
    const __bf16* wt = head ? Wv1T : Wp1T;
    const float*  bb = head ? b_v1 : b_p1;
#pragma unroll
    for (int nt = 0; nt < 4; nt++) {
      float bv = bb[nt * 16 + n];
      v8f a;
#pragma unroll
      for (int r = 0; r < 8; r++) a[r] = bv;
#pragma unroll
      for (int kc = 0; kc < 4; kc++)
        a = wmma_bf16(Af[kc],
                      make_bfrag_bf16(wt + (long)(nt * 16 + n) * HH, kc * 32, hi),
                      a);
#pragma unroll
      for (int r = 0; r < 8; r++)
        act[w][head][r + 8 * hi][nt * 16 + n] = ftanh(a[r]);
    }
  }

  // Stage 2: 16 rows x 9 cols per tile (per-wave, lockstep LDS ordering).
  for (int idx = lane; idx < 16 * (AA + 1); idx += 32) {
    int row = idx / (AA + 1), col = idx - row * (AA + 1);
    float s;
    if (col < AA) {
      s = b_p2[col];
#pragma unroll 8
      for (int k = 0; k < MHH; k++) s += act[w][0][row][k] * W_p2[k * AA + col];
    } else {
      s = b_v2[0];
#pragma unroll 8
      for (int k = 0; k < MHH; k++) s += act[w][1][row][k] * W_v2[k];
    }
    out[(grow0 + row) * (AA + 1) + col] = s;
  }
}

// ================== host-side launcher ==================
extern "C" void kernel_launch(void* const* d_in, const int* in_sizes, int n_in,
                              void* d_out, int out_size, void* d_ws, size_t ws_size,
                              hipStream_t stream) {
  (void)in_sizes; (void)n_in; (void)out_size; (void)ws_size;

  const float* x    = (const float*)d_in[0];
  const float* done = (const float*)d_in[1];
  const float* h0   = (const float*)d_in[2];
  const float* c0   = (const float*)d_in[3];
  const float* W_ih = (const float*)d_in[4];
  const float* W_hh = (const float*)d_in[5];
  const float* b_ih = (const float*)d_in[6];
  const float* b_hh = (const float*)d_in[7];
  const float* W_p1 = (const float*)d_in[8];
  const float* b_p1 = (const float*)d_in[9];
  const float* W_p2 = (const float*)d_in[10];
  const float* b_p2 = (const float*)d_in[11];
  const float* W_v1 = (const float*)d_in[12];
  const float* b_v1 = (const float*)d_in[13];
  const float* W_v2 = (const float*)d_in[14];
  const float* b_v2 = (const float*)d_in[15];

  float* out = (float*)d_out;                       // [T*B, A+1]
  float* hT  = out + TBROWS * (AA + 1);             // [B, H]
  float* cT  = hT + (long)BB * HH;                  // [B, H]

  // Workspace layout:
  //   Xg2   fp32 fragment blocks [T*B/16][32][256]          : 256 MB
  //   hseqB bf16 fragment blocks [T*B/16][8][256]           :  32 MB
  //   W_ihB bf16 [4H][F], W_hhB bf16 [4H][H],
  //   Wp1T/Wv1T bf16 [MH][H]                                : ~0.7 MB
  char* ws = (char*)d_ws;
  float*  Xg2   = (float*)ws;
  __bf16* hseqB = (__bf16*)(ws + (size_t)TBROWS * G4H * sizeof(float));
  __bf16* W_ihB = hseqB + (size_t)TBROWS * HH;
  __bf16* W_hhB = W_ihB + (size_t)G4H * FF;
  __bf16* Wp1T  = W_hhB + (size_t)G4H * HH;
  __bf16* Wv1T  = Wp1T  + (size_t)MHH * HH;

  prep_weights_kernel<<<256, 256, 0, stream>>>(W_ih, W_hh, W_p1, W_v1,
                                               W_ihB, W_hhB, Wp1T, Wv1T);
  gates_gemm_kernel<<<(int)(TBROWS / MR), 256, 0, stream>>>(
      x, W_ihB, b_ih, b_hh, Xg2);
  lstm_recurrent_kernel<<<BB / 16, 256, 0, stream>>>(
      Xg2, done, h0, c0, W_hhB, hseqB, hT, cT);
  heads_kernel<<<(int)(TBROWS / 16 / 8), 256, 0, stream>>>(
      hseqB, Wp1T, b_p1, W_p2, b_p2, Wv1T, b_v1, W_v2, b_v2, out);
}
